// TextRNN_14044543058505
// MI455X (gfx1250) — matrix operand order repaired and checked by reading, not compiled, not run on hardware
//
#include <hip/hip_runtime.h>
#include <hip/hip_bf16.h>
#include <math.h>

typedef _Float16 f16;
typedef __attribute__((ext_vector_type(8)))  _Float16 v8h;
typedef __attribute__((ext_vector_type(16))) _Float16 v16h;
typedef __attribute__((ext_vector_type(8)))  float    v8f;

#define VOCAB  50000
#define EMB    256
#define SEQ    512
#define BATCH  64
#define LABELS 10
#define HID    256
#define G4     1024            // 4*H
#define TB     32768           // SEQ*BATCH

static __device__ __forceinline__ v16h make_v16h(v8h lo, v8h hi) {
  return __builtin_shufflevector(lo, hi, 0,1,2,3,4,5,6,7,8,9,10,11,12,13,14,15);
}
static __device__ __forceinline__ float sigmoidf_(float x) {
  return 1.0f / (1.0f + __expf(-x));
}

// ---------------------------------------------------------------------------
// Phase 0: convert Wx/Wh (f32, [K=256][N=1024]) to f16, transposed to
// N-major [dir][n=1024][k=256] so WMMA B-operand loads are contiguous b128s.
// ---------------------------------------------------------------------------
__global__ void k_prep_weights(const float* __restrict__ Wx_f, const float* __restrict__ Wh_f,
                               const float* __restrict__ Wx_b, const float* __restrict__ Wh_b,
                               f16* __restrict__ WxT, f16* __restrict__ WhT) {
  int idx = blockIdx.x * blockDim.x + threadIdx.x;   // 0 .. 2*1024*256-1
  int dir = idx / (G4 * EMB);
  int rem = idx % (G4 * EMB);
  int n = rem / EMB;
  int k = rem % EMB;
  const float* wx = dir ? Wx_b : Wx_f;
  const float* wh = dir ? Wh_b : Wh_f;
  WxT[idx] = (f16)wx[k * G4 + n];
  WhT[idx] = (f16)wh[k * G4 + n];
}

// ---------------------------------------------------------------------------
// Phase 1: zx[dir][t*64+b][g] = embed(tokens)[b][t] @ Wx_dir + bias_dir
// GEMM M=32768, N=1024, K=256 via v_wmma_f32_16x16x32_f16.
// One wave per (dir, M-tile, group of 8 N-tiles). B tiles are loaded in
// batches of 4 ahead of the wmma quartet so loads overlap matrix ops.
// ---------------------------------------------------------------------------
__global__ __launch_bounds__(256) void
k_zx_gemm(const int* __restrict__ tokens, const float* __restrict__ emb,
          const f16* __restrict__ WxT, const float* __restrict__ b_f,
          const float* __restrict__ b_b, float* __restrict__ zx) {
  int tid  = threadIdx.x;
  int lane = tid & 31;
  int wave = tid >> 5;
  int wg   = blockIdx.x * (blockDim.x >> 5) + wave;  // global wave id, < 32768

  int dir = wg >> 14;          // 16384 waves per direction
  int rem = wg & 16383;
  int mt  = rem >> 3;          // 0..2047  (M tile)
  int ng  = rem & 7;           // 0..7     (group of 8 N tiles)

  int nlane = lane & 15;
  int koffA = (lane < 16) ? 0 : 8;
  int koffB = (lane < 16) ? 0 : 8;
  int mofs  = (lane < 16) ? 0 : 8;

  // A row for this lane: global row r = t*64+b (time-major zx layout)
  int mrow  = mt * 16 + nlane;
  int b     = mrow & 63;
  int t     = mrow >> 6;
  int token = tokens[b * SEQ + t];
  const float* arow = emb + (size_t)token * EMB;

  const float* bias = dir ? b_b : b_f;
  const f16*   wxt  = WxT + (size_t)dir * G4 * EMB;
  float*       zxd  = zx  + (size_t)dir * TB * G4;

  v8f acc[8];
  for (int i = 0; i < 8; ++i) {
    float bv = bias[(ng * 8 + i) * 16 + nlane];
    for (int r = 0; r < 8; ++r) acc[i][r] = bv;
  }

  for (int kk = 0; kk < 8; ++kk) {
    int ka = kk * 32 + koffA;
    v8h alo, ahi;
    for (int j = 0; j < 8; ++j) {
      alo[j] = (f16)arow[ka + j];
      ahi[j] = (f16)arow[ka + 16 + j];
    }
    v16h A = make_v16h(alo, ahi);
    int kb = kk * 32 + koffB;

    for (int half = 0; half < 2; ++half) {
      // batch the 4 B-tile loads ahead of the wmma quartet -> partial waits
      v16h Bt[4];
      for (int i = 0; i < 4; ++i) {
        int n = (ng * 8 + half * 4 + i) * 16 + nlane;
        const f16* bp = wxt + (size_t)n * EMB + kb;
        Bt[i] = make_v16h(*(const v8h*)bp, *(const v8h*)(bp + 16));
      }
      for (int i = 0; i < 4; ++i) {
        acc[half * 4 + i] = __builtin_amdgcn_wmma_f32_16x16x32_f16(
            false, A, false, Bt[i], (short)0, acc[half * 4 + i], false, false);
      }
    }
  }

  for (int i = 0; i < 8; ++i) {
    int ncol = (ng * 8 + i) * 16 + nlane;
    for (int r = 0; r < 8; ++r) {
      int m = mt * 16 + r + mofs;
      zxd[(size_t)m * G4 + ncol] = acc[i][r];
    }
  }
}

// ---------------------------------------------------------------------------
// Phase 2: sequential LSTM scan. One 1024-thread (32-wave) persistent
// workgroup per direction. Per step: z = h_prev @ Wh + zx[t]; gates; c,h.
// h double-buffered in LDS as f16 (one barrier per step); c lives in VGPRs.
// Per K-step, all 4 gate B-tiles are loaded before the wmma quartet so the
// (latency-critical) load->wmma chain pipelines with partial loadcnt waits.
// ---------------------------------------------------------------------------
__global__ __launch_bounds__(1024) void
k_lstm(const f16* __restrict__ WhT, const float* __restrict__ zx,
       float* __restrict__ hfin) {
  __shared__ __align__(16) f16 h_sh[2][BATCH * HID];  // 2 x 32KB

  int tid  = threadIdx.x;
  int lane = tid & 31;
  int wave = tid >> 5;
  int dir  = blockIdx.x;

  const f16*   wht = WhT + (size_t)dir * G4 * HID;
  const float* zxd = zx  + (size_t)dir * TB * G4;

  int mt = wave >> 3;          // 0..3   batch tile (16 rows)
  int jp = wave & 7;           // handles hidden tiles jt = 2*jp, 2*jp+1

  int nlane = lane & 15;
  int koffA = (lane < 16) ? 0 : 8;
  int koffB = (lane < 16) ? 0 : 8;
  int mofs  = (lane < 16) ? 0 : 8;

  for (int i = tid; i < BATCH * HID; i += 1024) h_sh[0][i] = (f16)0.0f;

  v8f cst[2];
  for (int r = 0; r < 8; ++r) { cst[0][r] = 0.0f; cst[1][r] = 0.0f; }

  __syncthreads();

  for (int s = 0; s < SEQ; ++s) {
    int t   = dir ? (SEQ - 1 - s) : s;
    int cur = s & 1;
    const float* zrow = zxd + (size_t)t * BATCH * G4;

    if (s + 1 < SEQ) {  // warm caches for next step's zx slab
      int tn = dir ? (SEQ - 2 - s) : (s + 1);
      __builtin_prefetch(zxd + (size_t)tn * BATCH * G4 + (size_t)tid * 16, 0, 1);
    }

    for (int ht = 0; ht < 2; ++ht) {
      int jt   = jp * 2 + ht;
      int jcol = jt * 16 + nlane;

      v8f acc[4];
      for (int g = 0; g < 4; ++g)
        for (int r = 0; r < 8; ++r)
          acc[g][r] = zrow[(size_t)(mt * 16 + r + mofs) * G4 + g * HID + jcol];

      for (int kk = 0; kk < 8; ++kk) {
        int ka = kk * 32 + koffA;
        const f16* ap = &h_sh[cur][(mt * 16 + nlane) * HID + ka];
        v16h A = make_v16h(*(const v8h*)ap, *(const v8h*)(ap + 16));
        int kb = kk * 32 + koffB;

        v16h Bt[4];
        for (int g = 0; g < 4; ++g) {
          const f16* bp = wht + (size_t)(g * HID + jcol) * HID + kb;
          Bt[g] = make_v16h(*(const v8h*)bp, *(const v8h*)(bp + 16));
        }
        for (int g = 0; g < 4; ++g) {
          acc[g] = __builtin_amdgcn_wmma_f32_16x16x32_f16(
              false, A, false, Bt[g], (short)0, acc[g], false, false);
        }
      }

      // gates + state update; write new h into the other LDS buffer
      for (int r = 0; r < 8; ++r) {
        float iv = sigmoidf_(acc[0][r]);
        float fv = sigmoidf_(acc[1][r]);
        float gv = tanhf(acc[2][r]);
        float ov = sigmoidf_(acc[3][r]);
        float cv = fv * cst[ht][r] + iv * gv;
        cst[ht][r] = cv;
        float hv = ov * tanhf(cv);
        h_sh[cur ^ 1][(mt * 16 + r + mofs) * HID + jcol] = (f16)hv;
      }
    }
    __syncthreads();  // new-h visible before next step reads it
  }

  // final h lives in h_sh[(511+1)&1] == h_sh[0]
  for (int i = tid; i < BATCH * HID; i += 1024)
    hfin[(size_t)dir * BATCH * HID + i] = (float)h_sh[0][i];
}

// ---------------------------------------------------------------------------
// Phase 3: logits = [h_fwd, h_bwd] @ W_out + b_out, softmax. 64x10 outputs.
// ---------------------------------------------------------------------------
__global__ void k_out(const float* __restrict__ hfin, const float* __restrict__ W_out,
                      const float* __restrict__ b_out, float* __restrict__ out) {
  __shared__ float lg[LABELS];
  int b   = blockIdx.x;
  int tid = threadIdx.x;
  if (tid < LABELS) {
    const float* hf = hfin + (size_t)b * HID;
    const float* hb = hfin + (size_t)BATCH * HID + (size_t)b * HID;
    float s = b_out[tid];
    for (int k = 0; k < HID; ++k) s += hf[k] * W_out[k * LABELS + tid];
    for (int k = 0; k < HID; ++k) s += hb[k] * W_out[(HID + k) * LABELS + tid];
    lg[tid] = s;
  }
  __syncthreads();
  if (tid < LABELS) {
    float m = lg[0];
    for (int j = 1; j < LABELS; ++j) m = fmaxf(m, lg[j]);
    float sum = 0.0f;
    for (int j = 0; j < LABELS; ++j) sum += __expf(lg[j] - m);
    out[b * LABELS + tid] = __expf(lg[tid] - m) / sum;
  }
}

// ---------------------------------------------------------------------------
extern "C" void kernel_launch(void* const* d_in, const int* in_sizes, int n_in,
                              void* d_out, int out_size, void* d_ws, size_t ws_size,
                              hipStream_t stream) {
  const int*   tokens = (const int*)  d_in[0];
  const float* emb    = (const float*)d_in[1];
  const float* Wx_f   = (const float*)d_in[2];
  const float* Wh_f   = (const float*)d_in[3];
  const float* b_f    = (const float*)d_in[4];
  const float* Wx_b   = (const float*)d_in[5];
  const float* Wh_b   = (const float*)d_in[6];
  const float* b_b    = (const float*)d_in[7];
  const float* W_out  = (const float*)d_in[8];
  const float* b_out  = (const float*)d_in[9];
  float* out = (float*)d_out;

  // workspace layout
  char* ws = (char*)d_ws;
  f16*   WxT  = (f16*)ws;                                   // 2*1024*256*2 = 1 MB
  f16*   WhT  = (f16*)(ws + (size_t)(1 << 20));             // 1 MB
  float* zx   = (float*)(ws + (size_t)(2 << 20));           // 2*32768*1024*4 = 256 MB
  float* hfin = (float*)(ws + (size_t)(2 << 20) + (size_t)2 * TB * G4 * 4);  // 128 KB

  k_prep_weights<<<2 * G4 * EMB / 256, 256, 0, stream>>>(Wx_f, Wh_f, Wx_b, Wh_b, WxT, WhT);
  k_zx_gemm<<<4096, 256, 0, stream>>>(tokens, emb, WxT, b_f, b_b, zx);
  k_lstm<<<2, 1024, 0, stream>>>(WhT, zx, hfin);
  k_out<<<BATCH, 32, 0, stream>>>(hfin, W_out, b_out, out);
}
